// BatchGLACNet_64321430224988
// MI455X (gfx1250) — matrix operand order repaired and checked by reading, not compiled, not run on hardware
//
#include <hip/hip_runtime.h>
#include <hip/hip_bf16.h>
#include <math.h>

// ---------------- types ----------------
typedef __bf16 bf16_t;
typedef __attribute__((ext_vector_type(16))) __bf16 v16bf;
typedef __attribute__((ext_vector_type(8)))  __bf16 bf16x8;
typedef __attribute__((ext_vector_type(8)))  float   v8f;

#define B_    16
#define S_    5
#define T_    24
#define FIMG  2048
#define H_    512
#define HD_   512
#define E_    512
#define V_    20000
#define BN_SCALE 0.99999500003749975f   // 1/sqrt(1+1e-5)

// ---------------- WMMA fragment loaders (bf16, 16x16x32) ----------------
// A (16xK, M per lane = lane%16):  K(j) = (j/8)*16 + (lane/16)*8 + j%8
__device__ __forceinline__ v16bf load_a_frag(const bf16_t* base, int ld, int row,
                                             int kbase, int half) {
  const bf16_t* p = base + (long)row * ld + kbase + half * 8;
  bf16x8 lo = *(const bf16x8*)(p);
  bf16x8 hi = *(const bf16x8*)(p + 16);
  v16bf r;
#pragma unroll
  for (int j = 0; j < 8; ++j) { r[j] = lo[j]; r[j + 8] = hi[j]; }
  return r;
}
// B (Kx16, N per lane = lane%16):  K(j) = (j/8)*16 + (lane/16)*8 + j%8
__device__ __forceinline__ v16bf load_b_frag(const bf16_t* base, int ld, int row,
                                             int kbase, int half) {
  const bf16_t* p = base + (long)row * ld + kbase + half * 8;
  bf16x8 lo = *(const bf16x8*)(p);
  bf16x8 hi = *(const bf16x8*)(p + 16);
  v16bf r;
#pragma unroll
  for (int j = 0; j < 8; ++j) { r[j] = lo[j]; r[j + 8] = hi[j]; }
  return r;
}

__device__ __forceinline__ float sigm(float x) { return 1.f / (1.f + __expf(-x)); }

// ---------------- generic bf16 WMMA GEMM:  C = BN(A @ W^T + bias) ----------
// A [M,K] row-major bf16, W [N,K] row-major bf16 (torch [out,in] layout).
// Each wave computes an (MT*16)x16 tile: one B fragment is reused across MT
// A fragments per k-step, cutting W (L2) traffic MT-fold on the big GEMMs.
// remap_bst: output row (s*T+t)*B+b -> b*S*T + s*T + t   (logits into d_out).
template <int MT>
__global__ __launch_bounds__(128)
void gemm_bf16_wmma(const bf16_t* __restrict__ A, int lda,
                    const bf16_t* __restrict__ W, int ldw,
                    float* __restrict__ C, int ldc,
                    int M, int N, int K,
                    const float* __restrict__ bias,
                    const float* __restrict__ bn_g,
                    const float* __restrict__ bn_b,
                    int remap_bst) {
  const int lane = threadIdx.x & 31;
  const int wave = threadIdx.x >> 5;
  const int half = lane >> 4;
  const int mtile0 = blockIdx.x * MT;
  const int ntile = blockIdx.y * 4 + wave;       // wave-uniform
  if (ntile * 16 >= N) return;                   // uniform per wave -> EXEC stays full
  const int rowW = ntile * 16 + (lane & 15);
  v8f acc[MT] = {};
  for (int k = 0; k < K; k += 32) {
    v16bf b = load_b_frag(W, ldw, rowW, k, half);
#pragma unroll
    for (int mt = 0; mt < MT; ++mt) {
      v16bf a = load_a_frag(A, lda, (mtile0 + mt) * 16 + (lane & 15), k, half);
      acc[mt] = __builtin_amdgcn_wmma_f32_16x16x32_bf16(false, a, false, b, (short)0,
                                                        acc[mt], false, false);
    }
  }
  const int n = ntile * 16 + (lane & 15);
  const float bv = bias ? bias[n] : 0.f;
#pragma unroll
  for (int mt = 0; mt < MT; ++mt) {
#pragma unroll
    for (int r = 0; r < 8; ++r) {
      int m = (mtile0 + mt) * 16 + r + 8 * half;
      float v = acc[mt][r] + bv;
      if (bn_g) v = bn_g[n] * (v * BN_SCALE) + bn_b[n];
      long row = m;
      if (remap_bst) { int b = m % B_; int st = m / B_; row = (long)b * (S_ * T_) + st; }
      C[row * (long)ldc + n] = v;
    }
  }
}

// ---------------- elementwise helpers ----------------
__global__ void cvt_f32_bf16(const float* __restrict__ s, bf16_t* __restrict__ d, int n) {
  int i = blockIdx.x * blockDim.x + threadIdx.x;
  if (i < n) d[i] = (bf16_t)s[i];
}
__global__ void add_vec(const float* __restrict__ a, const float* __restrict__ b,
                        float* __restrict__ o, int n) {
  int i = blockIdx.x * blockDim.x + threadIdx.x;
  if (i < n) o[i] = a[i] + b[i];
}

// glocal = concat(local, hs_f(swapped), hs_r(swapped)) as bf16, rows m=b*S+s
__global__ void glocal_assemble(const bf16_t* __restrict__ local_bf,
                                const float* __restrict__ hs_f,
                                const float* __restrict__ hs_r,
                                bf16_t* __restrict__ glocal) {
  const int W = FIMG + 2 * H_;
  int idx = blockIdx.x * blockDim.x + threadIdx.x;
  if (idx >= B_ * S_ * W) return;
  int m = idx / W, c = idx % W;
  int b = m / S_, s = m % S_;
  bf16_t v;
  if (c < FIMG)            v = local_bf[(long)m * FIMG + c];
  else if (c < FIMG + H_)  v = (bf16_t)hs_f[((long)s * B_ + b) * H_ + (c - FIMG)];
  else                     v = (bf16_t)hs_r[((long)s * B_ + b) * H_ + (c - FIMG - H_)];
  glocal[idx] = v;
}

// teacher-forced embedded inputs: rows (s*T+t)*B+b ; t==0 -> zeros
__global__ void emb_gather(const int* __restrict__ tok, const float* __restrict__ embed,
                           bf16_t* __restrict__ emb_in) {
  int idx = blockIdx.x * blockDim.x + threadIdx.x;
  if (idx >= S_ * T_ * B_ * E_) return;
  int e = idx % E_; int row = idx / E_;
  int b = row % B_; int st = row / B_;
  int t = st % T_;  int s = st / T_;
  float v = 0.f;
  if (t > 0) {
    int token = tok[((long)b * S_ + s) * T_ + (t - 1)];
    v = embed[(long)token * E_ + e];
  }
  emb_in[idx] = (bf16_t)v;
}

// eos[s*B+b] = first t with token==pad_num-1 else 0  (torch.max/argmax semantics)
__global__ void eos_kernel(const int* __restrict__ tok, const int* __restrict__ padp,
                           int* __restrict__ eos) {
  int idx = blockIdx.x * blockDim.x + threadIdx.x;
  if (idx >= S_ * B_) return;
  int s = idx / B_, b = idx % B_;
  int pad = padp[0] - 1;
  int e = 0;
  for (int t = 0; t < T_; ++t)
    if (tok[((long)b * S_ + s) * T_ + t] == pad) { e = t; break; }
  eos[idx] = e;
}

// ---------------- encoder LSTM recurrence (one WG per direction) ------------
// gates = Xpre[row(b,s)] + h @ whh^T ; h kept bf16 in LDS, c in registers.
__global__ __launch_bounds__(256)
void enc_lstm_kernel(const float* __restrict__ Xpre_f, const float* __restrict__ Xpre_r,
                     const bf16_t* __restrict__ whh_f, const bf16_t* __restrict__ whh_r,
                     float* __restrict__ hs_f, float* __restrict__ hs_r) {
  const int dir = blockIdx.x;
  const float* Xpre = dir ? Xpre_r : Xpre_f;
  const bf16_t* whh = dir ? whh_r : whh_f;
  float* hs = dir ? hs_r : hs_f;

  __shared__ bf16_t h_bf[B_ * H_];
  __shared__ bf16_t h_nx[B_ * H_];
  for (int i = threadIdx.x; i < B_ * H_; i += blockDim.x) h_bf[i] = (bf16_t)0.f;
  __syncthreads();

  const int lane = threadIdx.x & 31;
  const int wave = threadIdx.x >> 5;
  const int half = lane >> 4;
  const int rA = lane & 15;

  float creg[4][8];
#pragma unroll
  for (int a = 0; a < 4; ++a)
#pragma unroll
    for (int r = 0; r < 8; ++r) creg[a][r] = 0.f;

  for (int stepi = 0; stepi < S_; ++stepi) {
    const int s = dir ? (S_ - 1 - stepi) : stepi;
#pragma unroll
    for (int jj = 0; jj < 4; ++jj) {
      const int j = wave * 4 + jj;                 // hid tile 0..31
      v8f acc[4] = {};
      for (int k = 0; k < H_; k += 32) {
        v16bf a = load_a_frag(h_bf, H_, rA, k, half);
#pragma unroll
        for (int g = 0; g < 4; ++g) {
          v16bf bb = load_b_frag(whh, H_, g * H_ + j * 16 + rA, k, half);
          acc[g] = __builtin_amdgcn_wmma_f32_16x16x32_bf16(false, a, false, bb,
                                                           (short)0, acc[g], false, false);
        }
      }
      const int n = j * 16 + rA;
#pragma unroll
      for (int r = 0; r < 8; ++r) {
        const int b = r + 8 * half;
        const float* xp = Xpre + ((long)(b * S_ + s)) * (4 * H_);
        float iv = acc[0][r] + xp[0 * H_ + n];
        float fv = acc[1][r] + xp[1 * H_ + n];
        float gv = acc[2][r] + xp[2 * H_ + n];
        float ov = acc[3][r] + xp[3 * H_ + n];
        float cn = sigm(fv) * creg[jj][r] + sigm(iv) * tanhf(gv);
        float hn = sigm(ov) * tanhf(cn);
        creg[jj][r] = cn;
        h_nx[b * H_ + n] = (bf16_t)hn;
        hs[((long)s * B_ + b) * H_ + n] = hn;
      }
    }
    __syncthreads();
    for (int i = threadIdx.x; i < B_ * H_; i += blockDim.x) h_bf[i] = h_nx[i];
    __syncthreads();
  }
}

// ---------------- decoder LSTM recurrence (single WG) -----------------------
__global__ __launch_bounds__(256)
void dec_lstm_kernel(const float* __restrict__ Xemb,   // [S*T*B, 2048] (+bias)
                     const float* __restrict__ Ximg,   // [B*S, 2048]
                     const bf16_t* __restrict__ whh,   // [2048, 512] bf16
                     const int* __restrict__ eos,      // [S*B]
                     float* __restrict__ Hall, float* __restrict__ Call,
                     bf16_t* __restrict__ Hrelu) {
  __shared__ bf16_t h_bf[B_ * HD_];
  __shared__ bf16_t h_nx[B_ * HD_];
  for (int i = threadIdx.x; i < B_ * HD_; i += blockDim.x) h_bf[i] = (bf16_t)0.f;
  __syncthreads();

  const int lane = threadIdx.x & 31;
  const int wave = threadIdx.x >> 5;
  const int half = lane >> 4;
  const int rA = lane & 15;

  float creg[4][8];
#pragma unroll
  for (int a = 0; a < 4; ++a)
#pragma unroll
    for (int r = 0; r < 8; ++r) creg[a][r] = 0.f;

  for (int s = 0; s < S_; ++s) {
    for (int t = 0; t < T_; ++t) {
      const int step = s * T_ + t;
#pragma unroll
      for (int jj = 0; jj < 4; ++jj) {
        const int j = wave * 4 + jj;
        v8f acc[4] = {};
        for (int k = 0; k < HD_; k += 32) {
          v16bf a = load_a_frag(h_bf, HD_, rA, k, half);
#pragma unroll
          for (int g = 0; g < 4; ++g) {
            v16bf bb = load_b_frag(whh, HD_, g * HD_ + j * 16 + rA, k, half);
            acc[g] = __builtin_amdgcn_wmma_f32_16x16x32_bf16(false, a, false, bb,
                                                             (short)0, acc[g], false, false);
          }
        }
        const int n = j * 16 + rA;
#pragma unroll
        for (int r = 0; r < 8; ++r) {
          const int b = r + 8 * half;
          const float* xe = Xemb + ((long)step * B_ + b) * (4 * HD_);
          const float* xi = Ximg + ((long)(b * S_ + s)) * (4 * HD_);
          float iv = acc[0][r] + xe[0 * HD_ + n] + xi[0 * HD_ + n];
          float fv = acc[1][r] + xe[1 * HD_ + n] + xi[1 * HD_ + n];
          float gv = acc[2][r] + xe[2 * HD_ + n] + xi[2 * HD_ + n];
          float ov = acc[3][r] + xe[3 * HD_ + n] + xi[3 * HD_ + n];
          float cn = sigm(fv) * creg[jj][r] + sigm(iv) * tanhf(gv);
          float hn = sigm(ov) * tanhf(cn);
          creg[jj][r] = cn;
          long row = (long)step * B_ + b;
          Hall[row * HD_ + n] = hn;
          Call[row * HD_ + n] = cn;
          Hrelu[row * HD_ + n] = (bf16_t)fmaxf(hn, 0.f);
          h_nx[b * HD_ + n] = (bf16_t)hn;
        }
      }
      __syncthreads();
      for (int i = threadIdx.x; i < B_ * HD_; i += blockDim.x) h_bf[i] = h_nx[i];
      __syncthreads();
    }
    // sentence carry: state at per-batch EOS step of this sentence
    __threadfence_block();
    __syncthreads();
    for (int i = threadIdx.x; i < B_ * HD_; i += blockDim.x) {
      int b = i / HD_;
      int e = eos[s * B_ + b];
      h_bf[i] = (bf16_t)Hall[((long)(s * T_ + e) * B_ + b) * HD_ + (i % HD_)];
    }
#pragma unroll
    for (int jj = 0; jj < 4; ++jj) {
      const int j = wave * 4 + jj;
      const int n = j * 16 + rA;
#pragma unroll
      for (int r = 0; r < 8; ++r) {
        const int b = r + 8 * half;
        int e = eos[s * B_ + b];
        creg[jj][r] = Call[((long)(s * T_ + e) * B_ + b) * HD_ + n];
      }
    }
    __syncthreads();
  }
}

// ---------------- row-wise log-softmax (in place on d_out) ------------------
__global__ __launch_bounds__(256)
void log_softmax_rows(float* __restrict__ out, int cols) {
  float* p = out + (long)blockIdx.x * cols;
  __shared__ float red[256];
  float mx = -3.4e38f;
  for (int i = threadIdx.x; i < cols; i += blockDim.x) mx = fmaxf(mx, p[i]);
  red[threadIdx.x] = mx;
  __syncthreads();
  for (int o = 128; o; o >>= 1) {
    if (threadIdx.x < o) red[threadIdx.x] = fmaxf(red[threadIdx.x], red[threadIdx.x + o]);
    __syncthreads();
  }
  mx = red[0];
  __syncthreads();
  float sm = 0.f;
  for (int i = threadIdx.x; i < cols; i += blockDim.x) sm += __expf(p[i] - mx);
  red[threadIdx.x] = sm;
  __syncthreads();
  for (int o = 128; o; o >>= 1) {
    if (threadIdx.x < o) red[threadIdx.x] += red[threadIdx.x + o];
    __syncthreads();
  }
  const float ls = __logf(red[0]);
  __syncthreads();
  for (int i = threadIdx.x; i < cols; i += blockDim.x) p[i] = p[i] - mx - ls;
}

// ---------------- host orchestration ----------------
extern "C" void kernel_launch(void* const* d_in, const int* in_sizes, int n_in,
                              void* d_out, int out_size, void* d_ws, size_t ws_size,
                              hipStream_t stream) {
  (void)in_sizes; (void)n_in; (void)out_size; (void)ws_size;
  const int*   tok       = (const int*)  d_in[0];
  const float* img_in    = (const float*)d_in[1];
  const int*   padp      = (const int*)  d_in[2];
  const float* cnn_w     = (const float*)d_in[3];
  const float* cnn_b     = (const float*)d_in[4];
  const float* cnn_bn_g  = (const float*)d_in[5];
  const float* cnn_bn_b  = (const float*)d_in[6];
  const float* enc_wih_f = (const float*)d_in[7];
  const float* enc_whh_f = (const float*)d_in[8];
  const float* enc_bih_f = (const float*)d_in[9];
  const float* enc_bhh_f = (const float*)d_in[10];
  const float* enc_wih_r = (const float*)d_in[11];
  const float* enc_whh_r = (const float*)d_in[12];
  const float* enc_bih_r = (const float*)d_in[13];
  const float* enc_bhh_r = (const float*)d_in[14];
  const float* enc_lin_w = (const float*)d_in[15];
  const float* enc_lin_b = (const float*)d_in[16];
  const float* enc_bn_g  = (const float*)d_in[17];
  const float* enc_bn_b  = (const float*)d_in[18];
  const float* embed     = (const float*)d_in[19];
  const float* dec_wih   = (const float*)d_in[20];
  const float* dec_whh   = (const float*)d_in[21];
  const float* dec_bih   = (const float*)d_in[22];
  const float* dec_bhh   = (const float*)d_in[23];
  const float* out_w     = (const float*)d_in[24];
  const float* out_b     = (const float*)d_in[25];
  float* out = (float*)d_out;

  char* wp = (char*)d_ws;
  auto alloc = [&](size_t bytes) -> void* {
    void* r = (void*)wp;
    wp += (bytes + 255) & ~(size_t)255;
    return r;
  };
  const int BS = B_ * S_;                      // 80
  const int ST = S_ * T_;                      // 120
  const int MD = ST * B_;                      // 1920

  bf16_t* imgsA      = (bf16_t*)alloc((size_t)BS * FIMG * 2);
  bf16_t* cnn_w_bf   = (bf16_t*)alloc((size_t)FIMG * FIMG * 2);
  float*  local_f    = (float*) alloc((size_t)BS * FIMG * 4);
  bf16_t* local_bf   = (bf16_t*)alloc((size_t)BS * FIMG * 2);
  bf16_t* wih_f_bf   = (bf16_t*)alloc((size_t)4 * H_ * FIMG * 2);
  bf16_t* wih_r_bf   = (bf16_t*)alloc((size_t)4 * H_ * FIMG * 2);
  bf16_t* whh_f_bf   = (bf16_t*)alloc((size_t)4 * H_ * H_ * 2);
  bf16_t* whh_r_bf   = (bf16_t*)alloc((size_t)4 * H_ * H_ * 2);
  float*  bias_ef    = (float*) alloc((size_t)4 * H_ * 4);
  float*  bias_er    = (float*) alloc((size_t)4 * H_ * 4);
  float*  bias_dec   = (float*) alloc((size_t)4 * HD_ * 4);
  float*  Xpre_f     = (float*) alloc((size_t)BS * 4 * H_ * 4);
  float*  Xpre_r     = (float*) alloc((size_t)BS * 4 * H_ * 4);
  float*  hs_f       = (float*) alloc((size_t)S_ * B_ * H_ * 4);
  float*  hs_r       = (float*) alloc((size_t)S_ * B_ * H_ * 4);
  bf16_t* glocal     = (bf16_t*)alloc((size_t)BS * (FIMG + 2 * H_) * 2);
  bf16_t* lin_w_bf   = (bf16_t*)alloc((size_t)2 * H_ * (2 * H_ + FIMG) * 2);
  float*  img_vecs   = (float*) alloc((size_t)BS * 2 * H_ * 4);
  bf16_t* img_bf     = (bf16_t*)alloc((size_t)BS * 2 * H_ * 2);
  bf16_t* dwih_bf    = (bf16_t*)alloc((size_t)4 * HD_ * (E_ + 2 * H_) * 2);
  bf16_t* dwhh_bf    = (bf16_t*)alloc((size_t)4 * HD_ * HD_ * 2);
  bf16_t* emb_in     = (bf16_t*)alloc((size_t)MD * E_ * 2);
  float*  Xemb       = (float*) alloc((size_t)MD * 4 * HD_ * 4);
  float*  Ximg       = (float*) alloc((size_t)BS * 4 * HD_ * 4);
  float*  Hall       = (float*) alloc((size_t)MD * HD_ * 4);
  float*  Call       = (float*) alloc((size_t)MD * HD_ * 4);
  bf16_t* Hrelu      = (bf16_t*)alloc((size_t)MD * HD_ * 2);
  bf16_t* out_w_bf   = (bf16_t*)alloc((size_t)V_ * HD_ * 2);
  int*    eos        = (int*)   alloc((size_t)S_ * B_ * 4);

  auto cvt = [&](const float* s, bf16_t* d, int n) {
    cvt_f32_bf16<<<(n + 255) / 256, 256, 0, stream>>>(s, d, n);
  };
  auto gemm = [&](const bf16_t* A, int lda, const bf16_t* W, int ldw,
                  float* C, int ldc, int M, int N, int K,
                  const float* bias, const float* g, const float* b2, int remap) {
    if (M % 64 == 0) {   // big GEMMs: 64x16 per wave, 4x B reuse
      dim3 grid(M / 64, (N / 16 + 3) / 4);
      gemm_bf16_wmma<4><<<grid, 128, 0, stream>>>(A, lda, W, ldw, C, ldc, M, N, K,
                                                  bias, g, b2, remap);
    } else {
      dim3 grid(M / 16, (N / 16 + 3) / 4);
      gemm_bf16_wmma<1><<<grid, 128, 0, stream>>>(A, lda, W, ldw, C, ldc, M, N, K,
                                                  bias, g, b2, remap);
    }
  };

  // weight / input conversions
  cvt(img_in, imgsA, BS * FIMG);
  cvt(cnn_w, cnn_w_bf, FIMG * FIMG);
  cvt(enc_wih_f, wih_f_bf, 4 * H_ * FIMG);
  cvt(enc_wih_r, wih_r_bf, 4 * H_ * FIMG);
  cvt(enc_whh_f, whh_f_bf, 4 * H_ * H_);
  cvt(enc_whh_r, whh_r_bf, 4 * H_ * H_);
  cvt(enc_lin_w, lin_w_bf, 2 * H_ * (2 * H_ + FIMG));
  cvt(dec_wih, dwih_bf, 4 * HD_ * (E_ + 2 * H_));
  cvt(dec_whh, dwhh_bf, 4 * HD_ * HD_);
  cvt(out_w, out_w_bf, V_ * HD_);
  add_vec<<<(4 * H_ + 255) / 256, 256, 0, stream>>>(enc_bih_f, enc_bhh_f, bias_ef, 4 * H_);
  add_vec<<<(4 * H_ + 255) / 256, 256, 0, stream>>>(enc_bih_r, enc_bhh_r, bias_er, 4 * H_);
  add_vec<<<(4 * HD_ + 255) / 256, 256, 0, stream>>>(dec_bih, dec_bhh, bias_dec, 4 * HD_);

  // 1) EncoderCNN: local = BN(imgs @ cnn_w^T + b)
  gemm(imgsA, FIMG, cnn_w_bf, FIMG, local_f, FIMG, BS, FIMG, FIMG,
       cnn_b, cnn_bn_g, cnn_bn_b, 0);
  cvt(local_f, local_bf, BS * FIMG);

  // 2) encoder LSTM input projections (bias = bih+bhh baked in)
  gemm(local_bf, FIMG, wih_f_bf, FIMG, Xpre_f, 4 * H_, BS, 4 * H_, FIMG,
       bias_ef, nullptr, nullptr, 0);
  gemm(local_bf, FIMG, wih_r_bf, FIMG, Xpre_r, 4 * H_, BS, 4 * H_, FIMG,
       bias_er, nullptr, nullptr, 0);

  // 3) bi-LSTM recurrence (2 workgroups: fwd / rev)
  enc_lstm_kernel<<<2, 256, 0, stream>>>(Xpre_f, Xpre_r, whh_f_bf, whh_r_bf, hs_f, hs_r);

  // 4) glocal concat + enc_lin + BN
  {
    int n = BS * (FIMG + 2 * H_);
    glocal_assemble<<<(n + 255) / 256, 256, 0, stream>>>(local_bf, hs_f, hs_r, glocal);
  }
  gemm(glocal, FIMG + 2 * H_, lin_w_bf, FIMG + 2 * H_, img_vecs, 2 * H_,
       BS, 2 * H_, FIMG + 2 * H_, enc_lin_b, enc_bn_g, enc_bn_b, 0);
  cvt(img_vecs, img_bf, BS * 2 * H_);

  // 5) decoder input projections
  gemm(img_bf, 2 * H_, dwih_bf + E_, E_ + 2 * H_, Ximg, 4 * HD_,
       BS, 4 * HD_, 2 * H_, nullptr, nullptr, nullptr, 0);
  {
    int n = MD * E_;
    emb_gather<<<(n + 255) / 256, 256, 0, stream>>>(tok, embed, emb_in);
    eos_kernel<<<1, 128, 0, stream>>>(tok, padp, eos);
  }
  gemm(emb_in, E_, dwih_bf, E_ + 2 * H_, Xemb, 4 * HD_,
       MD, 4 * HD_, E_, bias_dec, nullptr, nullptr, 0);

  // 6) decoder recurrence (stores all h; relu(h) in bf16 for the big GEMM)
  dec_lstm_kernel<<<1, 256, 0, stream>>>(Xemb, Ximg, dwhh_bf, eos, Hall, Call, Hrelu);

  // 7) logits = relu(h) @ out_w^T + out_b, remapped to [B,S,T,V] in d_out
  gemm(Hrelu, HD_, out_w_bf, HD_, out, V_, MD, V_, HD_,
       out_b, nullptr, nullptr, 1);

  // 8) row-wise log-softmax in place
  log_softmax_rows<<<MD, 256, 0, stream>>>(out, V_);
}
